// ResonanceLayer_22797686407496
// MI455X (gfx1250) — matrix operand order repaired and checked by reading, not compiled, not run on hardware
//
#include <hip/hip_runtime.h>
#include <hip/hip_bf16.h>

typedef _Float16 f16;
typedef _Float16 v16h __attribute__((ext_vector_type(16)));
typedef _Float16 v8h  __attribute__((ext_vector_type(8)));
typedef float    v8f  __attribute__((ext_vector_type(8)));
typedef unsigned int v4u __attribute__((ext_vector_type(4)));
typedef int      v8i  __attribute__((ext_vector_type(8)));
typedef int      v4i  __attribute__((ext_vector_type(4)));

#define N_COEFFS 1025
#define KC       1040          // padded cos (or sin) half of K
#define KTOT     2080          // cos half + sin half
#define NKT      65            // KTOT / 32
#define WINDOW   2048
#define N_FRAMES 32
#define N_RES    32
#define NEXPR    4
#define NS       32768
#define NFFT     32768         // complex FFT length for 65536-pt real FFT
#define PI_F     3.14159265358979f
#define TWO_PI_F 6.28318530717959f

static __device__ inline float2 f2(float a, float b) { float2 r; r.x = a; r.y = b; return r; }

// LDS byte offset of a __shared__ object (strip flat aperture via AS3 cast)
typedef __attribute__((address_space(3))) const void lds_cv;
__device__ inline unsigned lds_offset(const void* p) {
    return (unsigned)(unsigned long long)(lds_cv*)p;
}

// ---------------------------------------------------------------------------
// Tensor Data Mover: DMA a 32x256 f16 tile (row stride 2048 elems) into LDS.
// D# per CDNA5 ISA 8.3/8.4; 2-D tile so groups 2/3 are zero.
// ---------------------------------------------------------------------------
__device__ inline void tdm_load_B_tile(const f16* gsrc, f16* ltile) {
    unsigned long long ga = (unsigned long long)(uintptr_t)gsrc;
    v4u g0;
    g0.x = 1u;                                                   // count=1, user mode
    g0.y = lds_offset(ltile);                                    // lds_addr (bytes)
    g0.z = (unsigned)(ga & 0xFFFFFFFFu);                         // global_addr[31:0]
    g0.w = (unsigned)((ga >> 32) & 0x01FFFFFFu) | 0x80000000u;   // addr[56:32] | type=2
    v8i g1;
    g1[0] = 0x00010000;   // data_size=2 bytes (code 1)
    g1[1] = 0x08000000;   // tensor_dim0 = 2048  (bits 79:48 low half)
    g1[2] = 0x08200000;   // tensor_dim1 = 2080  (bits 111:80 low half)
    g1[3] = 0x01000000;   // tile_dim0  = 256    (bits 127:112)
    g1[4] = 0x00000020;   // tile_dim1  = 32
    g1[5] = 2048;         // tensor_dim0_stride[31:0]
    g1[6] = 0;
    g1[7] = 0;
    v4i zz = {0, 0, 0, 0};
#if defined(__clang_major__) && __clang_major__ >= 23
    v8i z8 = {0, 0, 0, 0, 0, 0, 0, 0};
    __builtin_amdgcn_tensor_load_to_lds(g0, g1, zz, zz, z8, 0);
#else
    __builtin_amdgcn_tensor_load_to_lds(g0, g1, zz, zz, 0);
#endif
}

// ---------------------------------------------------------------------------
// WMMA fragment helpers (CDNA5 ISA 7.12.2 layouts, wave32)
// ---------------------------------------------------------------------------
__device__ inline v16h load_a_frag(const f16* A, int ld, int mBase, int kBase) {
    // 16-bit A 16x32: lane<16 -> M=lane, K={0..7,16..23}; lane>=16 -> K={8..15,24..31}
    int lane = threadIdx.x & 31;
    int m    = mBase + (lane & 15);
    int kOff = kBase + ((lane & 16) ? 8 : 0);
    const f16* p = A + (size_t)m * ld + kOff;
    v8h lo = *(const v8h*)p;
    v8h hi = *(const v8h*)(p + 16);
    v16h f;
    for (int i = 0; i < 8; ++i) { f[i] = lo[i]; f[8 + i] = hi[i]; }
    return f;
}

__device__ inline v16h load_b_frag(const f16* B, int ld, int kBase, int nBase) {
    // 16-bit B 32x16: lane<16 -> N=lane, K={0..7,16..23}; lane>=16 -> K={8..15,24..31}
    int lane = threadIdx.x & 31;
    int kOff = kBase + ((lane & 16) ? 8 : 0);
    const f16* p = B + (size_t)kOff * ld + nBase + (lane & 15);
    v16h f;
    for (int i = 0; i < 8; ++i) { f[i] = p[(size_t)i * ld]; f[8 + i] = p[(size_t)(16 + i) * ld]; }
    return f;
}

__device__ inline void store_c_frag(float* C, int ld, int mBase, int nBase, v8f acc) {
    // 32-bit C/D 16x16: VGPR g: lane<16 -> (M=g, N=lane); lane>=16 -> (M=8+g, N=lane-16)
    int lane = threadIdx.x & 31;
    int n    = nBase + (lane & 15);
    int m0   = mBase + ((lane & 16) ? 8 : 0);
    for (int g = 0; g < 8; ++g) C[(size_t)(m0 + g) * ld + n] = acc[g];
}

// ---------------------------------------------------------------------------
// K1: routed = einsum('becf,cr->berf')  (M=32 r, N=128 f, K=16 c pad 32)
// ---------------------------------------------------------------------------
__global__ __launch_bounds__(256) void k_routed(const float* __restrict__ ctrl,
                                                const float* __restrict__ router,
                                                float* __restrict__ outB) {
    int be = blockIdx.x;  // b*4+e, 0..7
    __shared__ __attribute__((aligned(16))) f16 sA[32 * 32];
    __shared__ __attribute__((aligned(16))) f16 sB[32 * 128];
    int tid = threadIdx.x;
    for (int i = tid; i < 32 * 32; i += 256) {
        int r = i >> 5, c = i & 31;
        sA[i] = (c < 16) ? (f16)router[c * 32 + r] : (f16)0.f;  // A[r][c] = router[c][r]
    }
    for (int i = tid; i < 32 * 128; i += 256) {
        int c = i >> 7, f = i & 127;
        sB[i] = (c < 16) ? (f16)ctrl[(be * 16 + c) * 128 + f] : (f16)0.f;
    }
    __syncthreads();
    int wave  = tid >> 5;       // 8 waves
    int nBase = wave * 16;
    v16h bF = load_b_frag(sB, 128, 0, nBase);
    for (int mt = 0; mt < 2; ++mt) {
        v16h aF = load_a_frag(sA, 32, mt * 16, 0);
        v8f  c  = {0.f, 0.f, 0.f, 0.f, 0.f, 0.f, 0.f, 0.f};
        c = __builtin_amdgcn_wmma_f32_16x16x32_f16(false, aF, false, bF, (short)0, c, false, false);
        store_c_frag(outB + (size_t)be * 32 * 128, 128, mt * 16, nBase, c);
    }
}

// ---------------------------------------------------------------------------
// K2: A-matrix for the iDFT GEMM.  A[rex][j][k] (32 x 2080 f16 per (r,expr))
// ---------------------------------------------------------------------------
__global__ __launch_bounds__(256) void k_prepA(const float* __restrict__ amp,
                                               const float* __restrict__ phase,
                                               const float* __restrict__ decay,
                                               f16* __restrict__ A) {
    int rex = blockIdx.x;  // r*4+ex, 0..127
    int r = rex >> 2, ex = rex & 3;
    f16* Ab = A + (size_t)rex * 32 * KTOT;
    for (int k = threadIdx.x; k < KC; k += 256) {
        if (k < N_COEFFS) {
            int idx  = (r * N_COEFFS + k) * NEXPR + ex;
            float a  = fabsf(amp[idx]);
            float ph = tanhf(phase[idx]) * PI_F;
            float dc = 0.5f + (1.f / (1.f + __expf(-decay[idx]))) * 0.5f * 0.9f;
            float ld = __logf(dc + 1e-12f);
            float wk = (k == 0 || k == 1024) ? 1.f : 2.f;
            float sc = wk * (1.f / 2048.f);
            float cp = __cosf(ph) * sc, sp = __sinf(ph) * sc;
            for (int j = 0; j < 32; ++j) {
                float mag = a * __expf((float)(j + 1) * ld);
                Ab[j * KTOT + k]      = (f16)(mag * cp);
                Ab[j * KTOT + KC + k] = (f16)(mag * sp);
            }
        } else {
            for (int j = 0; j < 32; ++j) {
                Ab[j * KTOT + k]      = (f16)0.f;
                Ab[j * KTOT + KC + k] = (f16)0.f;
            }
        }
    }
}

// K3: shared DFT basis B[k][t] (2080 x 2048 f16): cos rows then -sin rows
__global__ __launch_bounds__(256) void k_prepB(f16* __restrict__ B) {
    int idx = blockIdx.x * 256 + threadIdx.x;
    if (idx >= KTOT * WINDOW) return;
    int k = idx / WINDOW, t = idx - k * WINDOW;
    float v = 0.f;
    if (k < N_COEFFS) {
        unsigned ph = ((unsigned)k * (unsigned)t) & 2047u;   // mod 2048 exact under u32 wrap
        v = __cosf(TWO_PI_F * (float)ph * (1.f / 2048.f));
    } else if (k >= KC && k < KC + N_COEFFS) {
        unsigned ph = ((unsigned)(k - KC) * (unsigned)t) & 2047u;
        v = -__sinf(TWO_PI_F * (float)ph * (1.f / 2048.f));
    }
    B[idx] = (f16)v;
}

// ---------------------------------------------------------------------------
// K4: frames[rex][j][t] = (A @ B)[j][t] * hann(t)   -- WMMA f16 tiled GEMM
// B tiles (32x256 f16) DMA'd into LDS by the Tensor Data Mover, double-buffered
// grid: (8 t-chunks, 128 rex), block 256 (8 waves, each owns 32 columns)
// ---------------------------------------------------------------------------
__global__ __launch_bounds__(256) void k_resgemm(const f16* __restrict__ A,
                                                 const f16* __restrict__ B,
                                                 float* __restrict__ frames) {
    __shared__ __attribute__((aligned(16))) f16 sBt[2][32 * 256];   // 2 x 16KB
    int chunk = blockIdx.x;   // 0..7 (256 cols each)
    int rex   = blockIdx.y;   // 0..127
    const f16* Ab = A + (size_t)rex * 32 * KTOT;
    int wave      = threadIdx.x >> 5;
    int chunkBase = chunk * 256;
    int nLoc      = wave * 32;
    v8f a00 = {0,0,0,0,0,0,0,0}, a01 = a00, a10 = a00, a11 = a00;

    if (wave == 0) tdm_load_B_tile(B + chunkBase, sBt[0]);          // prologue tile 0
    for (int kt = 0; kt < NKT; ++kt) {
        int buf = kt & 1;
        if (wave == 0) __builtin_amdgcn_s_wait_tensorcnt(0);        // tile[buf] landed
        __syncthreads();                                            // publish; prev reads done
        if (wave == 0 && kt + 1 < NKT)
            tdm_load_B_tile(B + (size_t)(kt + 1) * 32 * WINDOW + chunkBase, sBt[buf ^ 1]);
        int kBase = kt * 32;
        if (kt + 1 < NKT)
            __builtin_prefetch(Ab + (size_t)(threadIdx.x & 31) * KTOT + kBase + 32, 0, 1);
        v16h fa0 = load_a_frag(Ab, KTOT, 0,  kBase);
        v16h fa1 = load_a_frag(Ab, KTOT, 16, kBase);
        v16h fb0 = load_b_frag(sBt[buf], 256, 0, nLoc);
        v16h fb1 = load_b_frag(sBt[buf], 256, 0, nLoc + 16);
        a00 = __builtin_amdgcn_wmma_f32_16x16x32_f16(false, fa0, false, fb0, (short)0, a00, false, false);
        a01 = __builtin_amdgcn_wmma_f32_16x16x32_f16(false, fa0, false, fb1, (short)0, a01, false, false);
        a10 = __builtin_amdgcn_wmma_f32_16x16x32_f16(false, fa1, false, fb0, (short)0, a10, false, false);
        a11 = __builtin_amdgcn_wmma_f32_16x16x32_f16(false, fa1, false, fb1, (short)0, a11, false, false);
    }
    float* Fb = frames + (size_t)rex * 32 * WINDOW;
    int lane = threadIdx.x & 31;
    for (int mt = 0; mt < 2; ++mt) {
        for (int nt = 0; nt < 2; ++nt) {
            v8f acc = (mt == 0) ? (nt == 0 ? a00 : a01) : (nt == 0 ? a10 : a11);
            int t  = chunkBase + nLoc + nt * 16 + (lane & 15);
            float hann = 0.5f - 0.5f * __cosf(TWO_PI_F * (float)t * (1.f / 2048.f));
            int m0 = mt * 16 + ((lane & 16) ? 8 : 0);
            for (int g = 0; g < 8; ++g) Fb[(size_t)(m0 + g) * WINDOW + t] = acc[g] * hann;
        }
    }
}

// ---------------------------------------------------------------------------
// In-LDS radix-2 FFT, N=32768 complex (256 KB LDS, CDNA5 320KB WGP LDS).
// DIT, input must be bit-reversed (15 bits). sgn=-1 fwd, +1 inv (unscaled).
// ---------------------------------------------------------------------------
__device__ void fft_lds(float2* s, float sgn) {
    for (int half = 1; half < NFFT; half <<= 1) {
        float ang0 = sgn * (PI_F / (float)half);
        for (int i = threadIdx.x; i < (NFFT >> 1); i += blockDim.x) {
            int k   = i & (half - 1);
            int idx = ((i & ~(half - 1)) << 1) | k;
            float sn, cs;
            __sincosf(ang0 * (float)k, &sn, &cs);
            float2 a = s[idx];
            float2 b = s[idx + half];
            float tx = b.x * cs - b.y * sn;
            float ty = b.x * sn + b.y * cs;
            s[idx]        = f2(a.x + tx, a.y + ty);
            s[idx + half] = f2(a.x - tx, a.y - ty);
        }
        __syncthreads();
    }
}

__device__ inline float res_sample(const float* Fb, int s) {   // overlap-add on load
    int j = s >> 10, tau = s & 1023;
    float v = Fb[(size_t)j * WINDOW + tau];
    if (j > 0) v += Fb[(size_t)(j - 1) * WINDOW + 1024 + tau];
    return v;
}

// K5: res -> packed-real forward FFT + norm
__global__ __launch_bounds__(1024) void k_fft_res(const float* __restrict__ frames,
                                                  float2* __restrict__ ZR,
                                                  float* __restrict__ rnorm) {
    extern __shared__ float2 smem[];
    __shared__ float red[1024];
    int rex = blockIdx.x;
    const float* Fb = frames + (size_t)rex * 32 * WINDOW;
    int tid = threadIdx.x;
    float ss = 0.f;
    for (int i = 0; i < 32; ++i) {
        int n = tid + i * 1024;
        float2 z = f2(0.f, 0.f);
        if (n < 16384) {  // real signal zero-padded to 65536
            z.x = res_sample(Fb, 2 * n);
            z.y = res_sample(Fb, 2 * n + 1);
            ss += z.x * z.x + z.y * z.y;
        }
        smem[__brev((unsigned)n) >> 17] = z;
    }
    red[tid] = ss;
    __syncthreads();
    for (int s = 512; s > 0; s >>= 1) { if (tid < s) red[tid] += red[tid + s]; __syncthreads(); }
    if (tid == 0) rnorm[rex] = 1.f / (sqrtf(red[0]) + 1e-8f);
    fft_lds(smem, -1.f);
    for (int i = 0; i < 32; ++i) { int n = tid + i * 1024; ZR[(size_t)rex * NFFT + n] = smem[n]; }
}

__device__ inline float noise_val(unsigned idx) {   // deterministic stand-in for jax PRNG
    unsigned s = idx * 747796405u + 2891336453u;
    unsigned w = ((s >> ((s >> 28) + 4u)) ^ s) * 277803737u;
    w = (w >> 22) ^ w;
    return (float)w * (2.0f / 4294967296.0f) - 1.0f;
}

__device__ inline float u_sample(const float* rb, unsigned gid, int t) {  // upsample 128->32768 * noise
    float pos = ((float)t + 0.5f) * (1.f / 256.f) - 0.5f;
    pos = fminf(fmaxf(pos, 0.f), 127.f);
    int lo = (int)pos;
    int hi = lo + 1 < 127 ? lo + 1 : 127;
    float w = pos - (float)lo;
    float v = rb[lo] * (1.f - w) + rb[hi] * w;
    return v * noise_val(gid * 32768u + (unsigned)t);
}

// K6: routed_up*noise -> packed-real forward FFT
__global__ __launch_bounds__(1024) void k_fft_u(const float* __restrict__ routedAll,
                                                float2* __restrict__ ZU) {
    extern __shared__ float2 smem[];
    unsigned gid = blockIdx.x;  // (b*4+e)*32+r
    const float* rb = routedAll + (size_t)gid * 128;
    int tid = threadIdx.x;
    for (int i = 0; i < 32; ++i) {
        int n = tid + i * 1024;
        float2 z = f2(0.f, 0.f);
        if (n < 16384) { z.x = u_sample(rb, gid, 2 * n); z.y = u_sample(rb, gid, 2 * n + 1); }
        smem[__brev((unsigned)n) >> 17] = z;
    }
    __syncthreads();
    fft_lds(smem, -1.f);
    for (int i = 0; i < 32; ++i) { int n = tid + i * 1024; ZU[(size_t)gid * NFFT + n] = smem[n]; }
}

// real-FFT untangle / repack helpers
__device__ inline float2 untangle(float2 z, float2 zm, float c, float s) {
    float2 A = f2(0.5f * (z.x + zm.x), 0.5f * (z.y - zm.y));
    float2 B = f2(0.5f * (z.x - zm.x), 0.5f * (z.y + zm.y));
    float2 t = f2(B.x * c - B.y * s, B.x * s + B.y * c);
    return f2(A.x + t.y, A.y - t.x);
}
__device__ inline float2 retangle(float2 p, float2 pm, float c, float s) {
    float2 A = f2(0.5f * (p.x + pm.x), 0.5f * (p.y - pm.y));
    float2 B = f2(0.5f * (p.x - pm.x), 0.5f * (p.y + pm.y));
    float2 t = f2(B.x * c - B.y * s, B.x * s + B.y * c);
    return f2(A.x - t.y, A.y + t.x);
}

__device__ inline float softmax4(const float* dp, int f, int ex) {
    float v0 = dp[f] + 1.f, v1 = dp[128 + f], v2 = dp[256 + f], v3 = dp[384 + f];
    float m = fmaxf(fmaxf(v0, v1), fmaxf(v2, v3));
    float e0 = __expf(v0 - m), e1 = __expf(v1 - m), e2 = __expf(v2 - m), e3 = __expf(v3 - m);
    float ev = ex == 0 ? e0 : ex == 1 ? e1 : ex == 2 ? e2 : e3;
    return ev / (e0 + e1 + e2 + e3);
}
__device__ inline float dweight(const float* deform, int be, int ex, int t) {
    float pos = ((float)t + 0.5f) * (1.f / 256.f) - 0.5f;
    pos = fminf(fmaxf(pos, 0.f), 127.f);
    int lo = (int)pos;
    int hi = lo + 1 < 127 ? lo + 1 : 127;
    float w = pos - (float)lo;
    const float* dp = deform + (size_t)be * 4 * 128;
    return (1.f - w) * softmax4(dp, lo, ex) + w * softmax4(dp, hi, ex);
}

// ---------------------------------------------------------------------------
// K7: per (b,e,r): loop expr { spectral multiply (x 1/||res||), inverse FFT
// in LDS, accumulate dweight*conv }, then X = tanh(x*|gain|)
// ---------------------------------------------------------------------------
__global__ __launch_bounds__(1024) void k_conv_ifft(const float2* __restrict__ ZU,
                                                    const float2* __restrict__ ZR,
                                                    const float* __restrict__ rnorm,
                                                    const float* __restrict__ deform,
                                                    const float* __restrict__ gains,
                                                    float* __restrict__ X) {
    extern __shared__ float2 smem[];
    int gid = blockIdx.x;        // be*32 + r
    int r = gid & 31, be = gid >> 5;
    int tid = threadIdx.x;
    float acc[32];
    for (int i = 0; i < 32; ++i) acc[i] = 0.f;
    const float2* Zu = ZU + (size_t)gid * NFFT;
    for (int ex = 0; ex < NEXPR; ++ex) {
        const float2* Zr = ZR + (size_t)(r * 4 + ex) * NFFT;
        float rn = rnorm[r * 4 + ex];
        for (int k = tid; k <= 16384; k += 1024) {
            int km = (NFFT - k) & (NFFT - 1);
            float2 zu = Zu[k], zum = Zu[km];
            float2 zr = Zr[k], zrm = Zr[km];
            float th = PI_F * (float)k * (1.f / (float)NFFT);
            float ct, st;
            __sincosf(th, &st, &ct);
            float2 Xu  = untangle(zu,  zum,  ct, -st);
            float2 Xum = untangle(zum, zu,  -ct, -st);
            float2 Xr  = untangle(zr,  zrm,  ct, -st);
            float2 Xrm = untangle(zrm, zr,  -ct, -st);
            float2 P1 = f2((Xu.x * Xr.x - Xu.y * Xr.y) * rn, (Xu.x * Xr.y + Xu.y * Xr.x) * rn);
            float2 P2 = f2((Xum.x * Xrm.x - Xum.y * Xrm.y) * rn, (Xum.x * Xrm.y + Xum.y * Xrm.x) * rn);
            smem[__brev((unsigned)k) >> 17] = retangle(P1, P2, ct, st);
            if (k > 0 && k < 16384)
                smem[__brev((unsigned)(NFFT - k)) >> 17] = retangle(P2, P1, -ct, st);
        }
        __syncthreads();
        fft_lds(smem, 1.f);     // ends with __syncthreads
        for (int i = 0; i < 32; ++i) {
            int t = tid * 32 + i;
            float2 wv = smem[t >> 1];
            float cv = ((t & 1) ? wv.y : wv.x) * (1.f / (float)NFFT);
            acc[i] += dweight(deform, be, ex, t) * cv;
        }
        __syncthreads();
    }
    float ga = fabsf(gains[r]);
    float* Xb = X + (size_t)gid * NS;
    for (int i = 0; i < 32; ++i) { int t = tid * 32 + i; Xb[t] = tanhf(acc[i] * ga); }
}

// K8: summed[be][t] = sum_r X[be][r][t] (fixed order -> deterministic)
__global__ __launch_bounds__(256) void k_sum(const float* __restrict__ X, float* __restrict__ out) {
    int idx = blockIdx.x * 256 + threadIdx.x;
    if (idx >= 8 * NS) return;
    int be = idx >> 15, t = idx & (NS - 1);
    float s = 0.f;
    for (int rr = 0; rr < N_RES; ++rr) s += X[((size_t)(be * 32 + rr)) * NS + t];
    out[idx] = s;
}

// ---------------------------------------------------------------------------
extern "C" void kernel_launch(void* const* d_in, const int* in_sizes, int n_in,
                              void* d_out, int out_size, void* d_ws, size_t ws_size,
                              hipStream_t stream) {
    const float* ctrl   = (const float*)d_in[0];
    const float* deform = (const float*)d_in[1];
    const float* router = (const float*)d_in[2];
    const float* amp    = (const float*)d_in[3];
    const float* phase  = (const float*)d_in[4];
    const float* decay  = (const float*)d_in[5];
    const float* gains  = (const float*)d_in[6];
    float* outSummed = (float*)d_out;               // (2,4,1,32768)
    float* outBefore = (float*)d_out + 8 * NS;      // (2,4,32,128)

    // workspace layout (bytes, 16B aligned)
    char* ws = (char*)d_ws;
    const size_t A_OFF  = 0;                               // 128*32*2080 f16
    const size_t B_OFF  = A_OFF  + (size_t)128 * 32 * KTOT * 2;
    const size_t F_OFF  = B_OFF  + (size_t)KTOT * WINDOW * 2;
    const size_t RN_OFF = F_OFF  + (size_t)128 * 32 * WINDOW * 4;
    const size_t ZR_OFF = RN_OFF + 512;
    const size_t ZU_OFF = ZR_OFF + (size_t)128 * NFFT * 8;
    const size_t X_OFF  = ZU_OFF + (size_t)256 * NFFT * 8;
    const size_t NEED   = X_OFF  + (size_t)256 * NS * 4;
    if (ws_size < NEED) return;
    f16*    wsA  = (f16*)(ws + A_OFF);
    f16*    wsB  = (f16*)(ws + B_OFF);
    float*  wsF  = (float*)(ws + F_OFF);
    float*  wsRN = (float*)(ws + RN_OFF);
    float2* wsZR = (float2*)(ws + ZR_OFF);
    float2* wsZU = (float2*)(ws + ZU_OFF);
    float*  wsX  = (float*)(ws + X_OFF);

    const size_t fftLds = (size_t)NFFT * sizeof(float2);   // 256 KB, fits 320KB WGP LDS

    k_routed<<<8, 256, 0, stream>>>(ctrl, router, outBefore);
    k_prepA<<<128, 256, 0, stream>>>(amp, phase, decay, wsA);
    k_prepB<<<(KTOT * WINDOW + 255) / 256, 256, 0, stream>>>(wsB);
    dim3 gGemm(8, 128);
    k_resgemm<<<gGemm, 256, 0, stream>>>(wsA, wsB, wsF);
    k_fft_res<<<128, 1024, fftLds, stream>>>(wsF, wsZR, wsRN);
    k_fft_u<<<256, 1024, fftLds, stream>>>(outBefore, wsZU);
    k_conv_ifft<<<256, 1024, fftLds, stream>>>(wsZU, wsZR, wsRN, deform, gains, wsX);
    k_sum<<<(8 * NS + 255) / 256, 256, 0, stream>>>(wsX, outSummed);
}
